// MAMBAClassifier_2654289789650
// MI455X (gfx1250) — compile-verified, matrix operand order checked by reading, not run on hardware
//
#include <hip/hip_runtime.h>
#include <hip/hip_bf16.h>

// ---------------- model dims ----------------
#define D_MODEL   256
#define D_STATE   16
#define N_CLASSES 10
#define N_LAYER   6
#define D_CONVW   4
#define IN_SIZE   12
#define D_INNER   512
#define DT_RANK   16
#define B_SZ      16
#define SEQ       512
#define ROWS      (B_SZ * SEQ)        // 8192

// ---------------- WMMA types ----------------
typedef __bf16 v16bf __attribute__((ext_vector_type(16)));
typedef float  v8f   __attribute__((ext_vector_type(8)));
typedef unsigned int u32x4 __attribute__((ext_vector_type(4)));

union Frag16 { u32x4 u[2]; v16bf v; };

// ============================================================
// f32 [L][K,N] -> bf16 [L][N,K] transposed convert (weights, one-time)
// Enumerates outputs for coalesced writes.
// ============================================================
__global__ void k_cvt_transpose(const float* __restrict__ in, __bf16* __restrict__ out,
                                int K, int N, int total) {
    int t = blockIdx.x * blockDim.x + threadIdx.x;
    if (t >= total) return;
    int per   = K * N;
    int layer = t / per;
    int r     = t - layer * per;
    int n     = r / K;
    int k     = r - n * K;
    out[t] = (__bf16)in[(size_t)layer * per + (size_t)k * N + n];
}

// ============================================================
// fc_in: h[8192,256] = x[8192,12] @ W[12,256] + b
// ============================================================
__global__ void __launch_bounds__(256) k_fc_in(const float* __restrict__ x,
                                               const float* __restrict__ W,
                                               const float* __restrict__ b,
                                               float* __restrict__ h) {
    int t = blockIdx.x * blockDim.x + threadIdx.x;
    int row = t >> 8;
    int col = t & 255;
    float acc = b[col];
#pragma unroll
    for (int j = 0; j < IN_SIZE; ++j)
        acc += x[row * IN_SIZE + j] * W[j * D_MODEL + col];
    h[t] = acc;
}

// ============================================================
// rmsnorm of each row (d=256) -> bf16
// ============================================================
__global__ void __launch_bounds__(256) k_rmsnorm_bf16(const float* __restrict__ h,
                                                      const float* __restrict__ w,
                                                      __bf16* __restrict__ out) {
    __shared__ float red[256];
    int row = blockIdx.x;
    int t = threadIdx.x;
    float v = h[(size_t)row * D_MODEL + t];
    red[t] = v * v;
    __syncthreads();
    for (int s = 128; s > 0; s >>= 1) {
        if (t < s) red[t] += red[t + s];
        __syncthreads();
    }
    float scale = rsqrtf(red[0] * (1.0f / D_MODEL) + 1e-5f);
    out[(size_t)row * D_MODEL + t] = (__bf16)(v * scale * w[t]);
}

// ============================================================
// bf16 WMMA GEMM, TN form: D[M,N] = A[M,K] * Bt[N,K]^T (+ Cin)
// A row-major [M,K]; Bt row-major [N,K] (pre-transposed weights).
// Fully static shape: K, NT (N-tiles per block), HAS_C are template
// params -> no runtime branches in the hot loop, accumulators stay
// pinned, EXEC is all-ones everywhere (WMMA requirement).
// Block = 128 threads = 4 waves; block tile = 64(M) x NT*16(N);
// each wave: one 16-row strip x NT N-tiles (A-fragment reuse).
// grid = (M/64, Ntiles/NT)
// ============================================================
template<int K, int NT, bool HAS_C>
__global__ void __launch_bounds__(128)
k_gemm_bf16_tn(const __bf16* __restrict__ A, const __bf16* __restrict__ Bt,
               const float* __restrict__ Cin, float* __restrict__ D, int ldc)
{
    const int lane = threadIdx.x & 31;
    const int wave = threadIdx.x >> 5;
    const int row0 = (blockIdx.x * 4 + wave) * 16;
    const int ncol0 = blockIdx.y * (NT * 16);

    const int mn   = lane & 15;                  // M row (A) / N row (Bt) / N col (C,D)
    const int koff = (lane & 16) ? 8 : 0;        // K-half select (ISA 16-bit operand layout)
    const int mc   = (lane & 16) ? 8 : 0;        // C/D M-offset (ISA 32-bit C layout)

    v8f c[NT];
#pragma unroll
    for (int nt = 0; nt < NT; ++nt) {
        if (HAS_C) {
#pragma unroll
            for (int r = 0; r < 8; ++r)
                c[nt][r] = Cin[(size_t)(row0 + mc + r) * ldc + ncol0 + nt * 16 + mn];
        } else {
#pragma unroll
            for (int r = 0; r < 8; ++r) c[nt][r] = 0.0f;
        }
    }

    const __bf16* arow = A  + (size_t)(row0 + mn) * K + koff;
    const __bf16* brow = Bt + (size_t)(ncol0 + mn) * K + koff;

#pragma unroll 2
    for (int k0 = 0; k0 < K; k0 += 32) {
        Frag16 af;
        af.u[0] = *(const u32x4*)(arow + k0);
        af.u[1] = *(const u32x4*)(arow + k0 + 16);
        if (k0 + 32 < K) {
            __builtin_prefetch(arow + k0 + 32, 0, 1);   // global_prefetch_b8 (static cond)
        }
#pragma unroll
        for (int nt = 0; nt < NT; ++nt) {
            Frag16 bf_;
            bf_.u[0] = *(const u32x4*)(brow + nt * 16 * K + k0);
            bf_.u[1] = *(const u32x4*)(brow + nt * 16 * K + k0 + 16);
            c[nt] = __builtin_amdgcn_wmma_f32_16x16x32_bf16(
                        /*neg_a=*/false, af.v, /*neg_b=*/false, bf_.v,
                        /*c_mod=*/(short)0, c[nt],
                        /*reuse_a=*/false, /*reuse_b=*/false);
        }
    }

#pragma unroll
    for (int nt = 0; nt < NT; ++nt) {
#pragma unroll
        for (int r = 0; r < 8; ++r)
            D[(size_t)(row0 + mc + r) * ldc + ncol0 + nt * 16 + mn] = c[nt][r];
    }
}

// ============================================================
// causal depthwise conv (width 4) + bias + SiLU
// input: y_inproj[row,1024] (xs = cols 0..511); outputs fp32 + bf16
// ============================================================
__global__ void __launch_bounds__(256) k_conv_silu(const float* __restrict__ yin,
                                                   const float* __restrict__ convW,
                                                   const float* __restrict__ convb,
                                                   float* __restrict__ xs_act,
                                                   __bf16* __restrict__ xs_bf) {
    int t = blockIdx.x * blockDim.x + threadIdx.x;   // ROWS * D_INNER
    int c = t & (D_INNER - 1);
    int row = t >> 9;
    int l = row & (SEQ - 1);
    float acc = convb[c];
#pragma unroll
    for (int j = 0; j < D_CONVW; ++j) {
        int li = l - (D_CONVW - 1) + j;
        if (li >= 0)
            acc += yin[(size_t)(row - (D_CONVW - 1) + j) * (2 * D_INNER) + c] * convW[c * D_CONVW + j];
    }
    float s = acc / (1.0f + __expf(-acc));           // SiLU
    xs_act[t] = s;
    xs_bf[t] = (__bf16)s;
}

// ============================================================
// delta = softplus(dt @ dt_W + dt_b) ; dt = xdbl[:, 0:16]
// ============================================================
__global__ void __launch_bounds__(256) k_dt_softplus(const float* __restrict__ xdbl,
                                                     const float* __restrict__ dtW,
                                                     const float* __restrict__ dtb,
                                                     float* __restrict__ delta) {
    int t = blockIdx.x * blockDim.x + threadIdx.x;   // ROWS * D_INNER
    int d = t & (D_INNER - 1);
    int row = t >> 9;
    float acc = dtb[d];
#pragma unroll
    for (int r = 0; r < DT_RANK; ++r)
        acc += xdbl[(size_t)row * 48 + r] * dtW[r * D_INNER + d];
    delta[t] = (acc > 20.0f) ? acc : log1pf(__expf(acc));
}

// ============================================================
// selective scan + gate: one block per batch, thread = channel d
// y_bf[row,d] = (scan_y + u*D) * silu(res)
// ============================================================
__global__ void __launch_bounds__(512) k_scan(const float* __restrict__ xs_act,
                                              const float* __restrict__ delta,
                                              const float* __restrict__ xdbl,
                                              const float* __restrict__ A_log,
                                              const float* __restrict__ Dpar,
                                              const float* __restrict__ yin,   // res at col 512..1023
                                              __bf16* __restrict__ ybf) {
    __shared__ float Bv[D_STATE];
    __shared__ float Cv[D_STATE];
    int b = blockIdx.x;
    int d = threadIdx.x;

    float Arow[D_STATE];
#pragma unroll
    for (int s = 0; s < D_STATE; ++s)
        Arow[s] = -__expf(A_log[d * D_STATE + s]);
    float Dd = Dpar[d];
    float hst[D_STATE];
#pragma unroll
    for (int s = 0; s < D_STATE; ++s) hst[s] = 0.0f;

    for (int l = 0; l < SEQ; ++l) {
        int row = b * SEQ + l;
        __syncthreads();
        if (d < D_STATE)           Bv[d]      = xdbl[(size_t)row * 48 + DT_RANK + d];
        else if (d < 2 * D_STATE)  Cv[d - 16] = xdbl[(size_t)row * 48 + DT_RANK + D_STATE + (d - 16)];
        __syncthreads();

        float u  = xs_act[(size_t)row * D_INNER + d];
        float dl = delta[(size_t)row * D_INNER + d];
        float dlu = dl * u;
        float y = 0.0f;
#pragma unroll
        for (int s = 0; s < D_STATE; ++s) {
            hst[s] = __expf(dl * Arow[s]) * hst[s] + dlu * Bv[s];
            y += hst[s] * Cv[s];
        }
        y += u * Dd;
        float r = yin[(size_t)row * (2 * D_INNER) + D_INNER + d];
        float sil = r / (1.0f + __expf(-r));
        ybf[(size_t)row * D_INNER + d] = (__bf16)(y * sil);
    }
}

// ============================================================
// final rmsnorm (last token only) + classification head
// ============================================================
__global__ void __launch_bounds__(256) k_head(const float* __restrict__ h,
                                              const float* __restrict__ wf,
                                              const float* __restrict__ fcW,
                                              const float* __restrict__ fcb,
                                              float* __restrict__ out) {
    __shared__ float red[256];
    __shared__ float xn[256];
    int b = blockIdx.x;
    int t = threadIdx.x;
    int row = b * SEQ + (SEQ - 1);
    float v = h[(size_t)row * D_MODEL + t];
    red[t] = v * v;
    __syncthreads();
    for (int s = 128; s > 0; s >>= 1) {
        if (t < s) red[t] += red[t + s];
        __syncthreads();
    }
    float scale = rsqrtf(red[0] * (1.0f / D_MODEL) + 1e-5f);
    xn[t] = v * scale * wf[t];
    __syncthreads();
    if (t < N_CLASSES) {
        float acc = fcb[t];
        for (int j = 0; j < D_MODEL; ++j)
            acc += xn[j] * fcW[j * N_CLASSES + t];
        out[b * N_CLASSES + t] = acc;
    }
}

// ============================================================
// host-side orchestration
// ============================================================
extern "C" void kernel_launch(void* const* d_in, const int* in_sizes, int n_in,
                              void* d_out, int out_size, void* d_ws, size_t ws_size,
                              hipStream_t stream) {
    const float* x         = (const float*)d_in[0];
    const float* fc_in_W   = (const float*)d_in[1];
    const float* fc_in_b   = (const float*)d_in[2];
    const float* norm_w    = (const float*)d_in[3];
    const float* in_proj_W = (const float*)d_in[4];
    const float* conv_W    = (const float*)d_in[5];
    const float* conv_b    = (const float*)d_in[6];
    const float* x_proj_W  = (const float*)d_in[7];
    const float* dt_proj_W = (const float*)d_in[8];
    const float* dt_proj_b = (const float*)d_in[9];
    const float* A_log     = (const float*)d_in[10];
    const float* D_par     = (const float*)d_in[11];
    const float* out_proj_W= (const float*)d_in[12];
    const float* norm_f_w  = (const float*)d_in[13];
    const float* fc_W      = (const float*)d_in[14];
    const float* fc_b      = (const float*)d_in[15];
    float* out = (float*)d_out;

    char* ws = (char*)d_ws;
    size_t off = 0;
    auto alloc = [&](size_t bytes) -> void* {
        void* p = ws + off;
        off += (bytes + 255) & ~(size_t)255;
        return p;
    };

    float*  h        = (float*) alloc((size_t)ROWS * D_MODEL * 4);
    __bf16* xn_bf    = (__bf16*)alloc((size_t)ROWS * D_MODEL * 2);
    float*  y_inproj = (float*) alloc((size_t)ROWS * 2 * D_INNER * 4);
    float*  xs_act   = (float*) alloc((size_t)ROWS * D_INNER * 4);
    __bf16* xs_bf    = (__bf16*)alloc((size_t)ROWS * D_INNER * 2);
    float*  xdbl     = (float*) alloc((size_t)ROWS * 48 * 4);
    float*  delta    = (float*) alloc((size_t)ROWS * D_INNER * 4);
    __bf16* y_bf     = (__bf16*)alloc((size_t)ROWS * D_INNER * 2);
    // transposed bf16 weights: [N,K] per layer
    __bf16* Wt_in    = (__bf16*)alloc((size_t)N_LAYER * 2 * D_INNER * D_MODEL * 2);
    __bf16* Wt_xp    = (__bf16*)alloc((size_t)N_LAYER * 48 * D_INNER * 2);
    __bf16* Wt_out   = (__bf16*)alloc((size_t)N_LAYER * D_MODEL * D_INNER * 2);

    // one-time weight transpose-conversions to bf16 [N,K]
    {
        int n1 = N_LAYER * D_MODEL * 2 * D_INNER;
        k_cvt_transpose<<<(n1 + 255) / 256, 256, 0, stream>>>(in_proj_W, Wt_in, D_MODEL, 2 * D_INNER, n1);
        int n2 = N_LAYER * D_INNER * 48;
        k_cvt_transpose<<<(n2 + 255) / 256, 256, 0, stream>>>(x_proj_W, Wt_xp, D_INNER, 48, n2);
        int n3 = N_LAYER * D_INNER * D_MODEL;
        k_cvt_transpose<<<(n3 + 255) / 256, 256, 0, stream>>>(out_proj_W, Wt_out, D_INNER, D_MODEL, n3);
    }

    // h = x @ fc_in_W + b
    k_fc_in<<<ROWS * D_MODEL / 256, 256, 0, stream>>>(x, fc_in_W, fc_in_b, h);

    const int MB = ROWS / 64;   // 128 blocks in M

    for (int i = 0; i < N_LAYER; ++i) {
        // rmsnorm -> bf16
        k_rmsnorm_bf16<<<ROWS, 256, 0, stream>>>(h, norm_w + i * D_MODEL, xn_bf);

        // in_proj: [8192,1024] = xn @ W (K=256), Ntiles=64 -> 16 y-blocks of NT=4
        {
            dim3 grid(MB, 16);
            k_gemm_bf16_tn<D_MODEL, 4, false><<<grid, 128, 0, stream>>>(
                xn_bf, Wt_in + (size_t)i * 2 * D_INNER * D_MODEL, nullptr, y_inproj,
                2 * D_INNER);
        }

        // causal conv + SiLU
        k_conv_silu<<<ROWS * D_INNER / 256, 256, 0, stream>>>(
            y_inproj, conv_W + (size_t)i * D_INNER * D_CONVW, conv_b + (size_t)i * D_INNER,
            xs_act, xs_bf);

        // x_proj: [8192,48] = xs @ W (K=512), Ntiles=3 -> 1 y-block of NT=3
        {
            dim3 grid(MB, 1);
            k_gemm_bf16_tn<D_INNER, 3, false><<<grid, 128, 0, stream>>>(
                xs_bf, Wt_xp + (size_t)i * 48 * D_INNER, nullptr, xdbl,
                48);
        }

        // delta = softplus(dt @ dtW + dtb)
        k_dt_softplus<<<ROWS * D_INNER / 256, 256, 0, stream>>>(
            xdbl, dt_proj_W + (size_t)i * DT_RANK * D_INNER, dt_proj_b + (size_t)i * D_INNER, delta);

        // selective scan + gate -> y_bf
        k_scan<<<B_SZ, D_INNER, 0, stream>>>(
            xs_act, delta, xdbl, A_log + (size_t)i * D_INNER * D_STATE,
            D_par + (size_t)i * D_INNER, y_inproj, y_bf);

        // out_proj fused residual: h = h + y_bf @ W (K=512), Ntiles=16 -> 4 y-blocks of NT=4
        {
            dim3 grid(MB, 4);
            k_gemm_bf16_tn<D_INNER, 4, true><<<grid, 128, 0, stream>>>(
                y_bf, Wt_out + (size_t)i * D_MODEL * D_INNER, h, h,
                D_MODEL);
        }
    }

    // final rmsnorm (last token) + head
    k_head<<<B_SZ, 256, 0, stream>>>(h, norm_f_w, fc_W, fc_b, out);
}
